// StandardCausalAttention_56367150793408
// MI455X (gfx1250) — matrix operand order repaired and checked by reading, not compiled, not run on hardware
//
#include <hip/hip_runtime.h>
#include <hip/hip_bf16.h>

// StandardCausalAttention for MI455X (gfx1250), wave32 + WMMA bf16.
//   x[2,2048,1024] f32, w_qkv[1024,3072], b_qkv[3072], w_out[1024,1024], b_out[1024]
// Pipeline: GEMM(x,Wqkv)+b -> flash attention (bf16 WMMA, f32 softmax) -> GEMM(.,Wout)+b
// Workspace: qkv (4096*3072 f32 = 50.3MB) + attn (4096*1024 f32 = 16.8MB) in d_ws.
// All index math is int32 (every buffer < 2^31 elements) so the backend can use
// 32-bit voffset/scale_offset global addressing instead of u64 chains.

typedef __attribute__((ext_vector_type(16))) __bf16 bf16x16;
typedef __attribute__((ext_vector_type(8)))  float  f32x8;

union Frag  { bf16x16 v; unsigned int u[8]; };
union Pack4 { __bf16 h[4]; uint2 u2; };

// ---- WMMA fragment loaders (CDNA5 ISA 7.12.2 layouts, wave32) ----------------
// A-matrix 16x32 bf16: lane m=lane&15, half=lane>>4.
//   elem j<8  -> K = k0 + half*8 + j ; elem j>=8 -> K = k0 + 16 + half*8 + (j-8)
__device__ inline bf16x16 load_a_frag(const __bf16* base, int row0, int ld, int k0) {
  const int lane = threadIdx.x & 31;
  const int m = lane & 15, half = lane >> 4;
  const unsigned int* p = reinterpret_cast<const unsigned int*>(base);
  Frag f;
  const int rb = (row0 + m) * ld + k0;
#pragma unroll
  for (int r = 0; r < 4; ++r)
    f.u[r] = p[(rb + half * 8 + 2 * r) >> 1];
#pragma unroll
  for (int r = 0; r < 4; ++r)
    f.u[4 + r] = p[(rb + 16 + half * 8 + 2 * r) >> 1];
  return f.v;
}

// B-matrix 32x16 bf16 from a [N][K]-contiguous (K-major per column) LDS buffer:
// lane n=lane&15 (+col0), half=lane>>4; elem j<8 -> K = k0 + half*8 + j ; elem j>=8 -> K = k0 + 16 + half*8 + (j-8).
__device__ inline bf16x16 load_b_frag(const __bf16* base, int col0, int ld, int k0) {
  const int lane = threadIdx.x & 31;
  const int n = lane & 15, half = lane >> 4;
  const unsigned int* p = reinterpret_cast<const unsigned int*>(base);
  Frag f;
  const int rb = (col0 + n) * ld + k0 + half * 8;
#pragma unroll
  for (int r = 0; r < 8; ++r)
    f.u[r] = p[(rb + 2 * (r & 3) + 16 * (r >> 2)) >> 1];
  return f.v;
}

__device__ inline f32x8 wmma_bf16(bf16x16 a, bf16x16 b, f32x8 c) {
  // (neg_a, A, neg_b, B, c_mod, C, reuse_a, reuse_b)
  return __builtin_amdgcn_wmma_f32_16x16x32_bf16(false, a, false, b, (short)0, c,
                                                 false, false);
}

__device__ inline Pack4 cvt4(float4 f) {
  Pack4 t;
  t.h[0] = (__bf16)f.x; t.h[1] = (__bf16)f.y;
  t.h[2] = (__bf16)f.z; t.h[3] = (__bf16)f.w;
  return t;
}

// ---- GEMM: C[M,N] = A[M,K] @ B[K,N] + bias[N], bf16 WMMA / f32 accumulate ----
// 256 threads = 8 waves; block tile 128(M) x 128(N), K-step 32, LDS ping-pong.
// Wave (wm=w&3 -> M, wn=w>>2 -> N) owns 32x64 = 2x4 WMMA accumulators.
__global__ __launch_bounds__(256) void gemm_bias_kernel(
    const float* __restrict__ A, const float* __restrict__ B,
    const float* __restrict__ bias, float* __restrict__ C,
    int M, int N, int K) {
  __shared__ __align__(16) __bf16 As[2][128 * 32];   // [m][k]
  __shared__ __align__(16) __bf16 Bs[2][128 * 32];   // [n][k] (transposed)

  const int tid = threadIdx.x;
  const int m0 = blockIdx.y * 128;
  const int n0 = blockIdx.x * 128;
  const int w  = tid >> 5;
  const int wm = (w & 3) * 32, wn = (w >> 2) * 64;

  // per-thread staging coordinates (constant across K-steps)
  const int r0 = tid >> 3, ca = (tid & 7) * 4;       // A: [128][32], 4 rows apart 32
  const int kb = tid >> 5, nb = (tid & 31) * 4;      // B: [32][128], 4 rows apart 8

  float4 ar[4], br[4];
  auto load_tiles = [&](int k0) {
#pragma unroll
    for (int i = 0; i < 4; ++i) {
      ar[i] = *reinterpret_cast<const float4*>(&A[(m0 + r0 + i * 32) * K + k0 + ca]);
      br[i] = *reinterpret_cast<const float4*>(&B[(k0 + kb + i * 8) * N + n0 + nb]);
    }
  };
  auto store_tiles = [&](int p) {
#pragma unroll
    for (int i = 0; i < 4; ++i) {
      *reinterpret_cast<uint2*>(&As[p][(r0 + i * 32) * 32 + ca]) = cvt4(ar[i]).u2;
      Pack4 t = cvt4(br[i]);
      __bf16* bp = &Bs[p][nb * 32 + kb + i * 8];     // scatter along n (rows of Bs)
      bp[0] = t.h[0]; bp[32] = t.h[1]; bp[64] = t.h[2]; bp[96] = t.h[3];
    }
  };

  f32x8 acc[2][4] = {};
  load_tiles(0);
  int p = 0;
  for (int k0 = 0; k0 < K; k0 += 32, p ^= 1) {
    store_tiles(p);
    const int kn = (k0 + 32 < K) ? (k0 + 32) : 0;    // uniform clamp (dummy reload)
    load_tiles(kn);                                  // next tile in flight
    __syncthreads();                                 // one barrier per K-step
    bf16x16 a0 = load_a_frag(As[p], wm, 32, 0);
    bf16x16 a1 = load_a_frag(As[p], wm + 16, 32, 0);
#pragma unroll
    for (int j = 0; j < 4; ++j) {
      bf16x16 bj = load_b_frag(Bs[p], wn + j * 16, 32, 0);
      acc[0][j] = wmma_bf16(a0, bj, acc[0][j]);
      acc[1][j] = wmma_bf16(a1, bj, acc[1][j]);
    }
  }

  // D layout: VGPR e, lane: row M = e + 8*(lane>>4), col N = lane&15.
  const int lane = tid & 31, n = lane & 15, half = lane >> 4;
  const int rbase = m0 + wm + 8 * half;
  const int cbase = n0 + wn + n;
#pragma unroll
  for (int j = 0; j < 4; ++j) {
    float bv = bias[cbase + j * 16];
#pragma unroll
    for (int i = 0; i < 2; ++i) {
      const int base = (rbase + i * 16) * N + cbase + j * 16;
#pragma unroll
      for (int e = 0; e < 8; ++e)
        C[base + e * N] = acc[i][j][e] + bv;
    }
  }
}

// ---- Flash attention (causal), one (b, h, 64-query tile) per block ----------
// qkv layout: [B*T, 3072]; q at col h*64, k at 1024+h*64, v at 2048+h*64.
// 128 threads = 4 waves; wave w owns q-rows q0+w*16 .. +15, full D=64 output.
// K/V tiles are LDS ping-pong double buffered (one barrier per 32-key tile).
__global__ __launch_bounds__(128) void attn_kernel(const float* __restrict__ qkv,
                                                   float* __restrict__ out) {
  const int q0 = blockIdx.x * 64;
  const int h = blockIdx.y, b = blockIdx.z;
  const int tid = threadIdx.x, w = tid >> 5, lane = tid & 31;
  const int n = lane & 15, half = lane >> 4;

  __shared__ __align__(16) __bf16 Qs[64 * 64];       // [qrow][d]
  __shared__ __align__(16) __bf16 Ks[2][32 * 64];    // [key][d]  (B-frag: n=key,K=d)
  __shared__ __align__(16) __bf16 Vts[2][64 * 32];   // [d][key]  (B-frag: n=d,K=key)
  __shared__ __align__(16) __bf16 Ps[4][16 * 32];    // per-wave P tile [m][key]
  __shared__ float Mst[4][16], Lst[4][16];

  const int rowbase = b * 2048 * 3072 + h * 64;

  // stage Q tile (batched b128 loads, packed b64 LDS stores)
  {
    float4 qr[8];
#pragma unroll
    for (int i = 0; i < 8; ++i) {                    // 1024 float4 / 128 threads
      int r = (tid >> 4) + i * 8, c = (tid & 15) * 4;
      qr[i] = *reinterpret_cast<const float4*>(&qkv[rowbase + (q0 + r) * 3072 + c]);
    }
#pragma unroll
    for (int i = 0; i < 8; ++i) {
      int r = (tid >> 4) + i * 8, c = (tid & 15) * 4;
      *reinterpret_cast<uint2*>(&Qs[r * 64 + c]) = cvt4(qr[i]).u2;
    }
  }
  if (lane < 16) { Mst[w][lane] = -3.0e38f; Lst[w][lane] = 0.0f; }

  // K/V staging coords (constant across tiles)
  const int kk0 = tid >> 4, cc = (tid & 15) * 4;
  float4 kr[4], vr[4];
  auto load_kv = [&](int ks) {
    // NOTE: last iteration over-reads one 32-row tile; it stays inside d_ws
    // (qkv is followed by the attn buffer) and the data is never consumed.
#pragma unroll
    for (int i = 0; i < 4; ++i) {
      int g = rowbase + (ks + kk0 + i * 8) * 3072 + cc;
      kr[i] = *reinterpret_cast<const float4*>(&qkv[g + 1024]);
      vr[i] = *reinterpret_cast<const float4*>(&qkv[g + 2048]);
    }
  };
  auto store_kv = [&](int p) {
#pragma unroll
    for (int i = 0; i < 4; ++i) {
      int kk = kk0 + i * 8;
      *reinterpret_cast<uint2*>(&Ks[p][kk * 64 + cc]) = cvt4(kr[i]).u2;
      Pack4 t = cvt4(vr[i]);
      __bf16* vp = &Vts[p][cc * 32 + kk];            // scatter along d (rows of Vts)
      vp[0] = t.h[0]; vp[32] = t.h[1]; vp[64] = t.h[2]; vp[96] = t.h[3];
    }
  };

  f32x8 o0 = {}, o1 = {}, o2 = {}, o3 = {};
  const int qmaxw = q0 + w * 16 + 15;
  const float scale = 0.125f;                        // 1/sqrt(64)

  load_kv(0);
  int p = 0;
  for (int ks = 0; ks <= q0 + 63; ks += 32, p ^= 1) {
    store_kv(p);
    load_kv(ks + 32);                                // next tile in flight
    __syncthreads();                                 // one barrier per tile

    if (ks > qmaxw) continue;                        // wave-uniform causal skip

    // S[16q x 32k] = Q(16x64) @ K^T, two 16x16 N-tiles, K-dim in 2 steps of 32
    f32x8 s0 = {}, s1 = {};
#pragma unroll
    for (int d0 = 0; d0 < 64; d0 += 32) {
      bf16x16 a = load_a_frag(Qs, w * 16, 64, d0);
      s0 = wmma_bf16(a, load_b_frag(Ks[p], 0, 64, d0), s0);
      s1 = wmma_bf16(a, load_b_frag(Ks[p], 16, 64, d0), s1);
    }

    // online softmax per q-row (row M = e + 8*half, key col = n across 16 lanes)
    float alpha[8];
#pragma unroll
    for (int e = 0; e < 8; ++e) {
      int row = e + 8 * half;
      int qg = q0 + w * 16 + row;
      float t0 = s0[e] * scale, t1 = s1[e] * scale;
      if (ks + n > qg)      t0 = -3.0e38f;           // causal mask
      if (ks + 16 + n > qg) t1 = -3.0e38f;
      float rmax = fmaxf(t0, t1);
#pragma unroll
      for (int msk = 8; msk; msk >>= 1)
        rmax = fmaxf(rmax, __shfl_xor(rmax, msk, 32));
      float mold = Mst[w][row];
      float lold = Lst[w][row];
      float mnew = fmaxf(mold, rmax);
      float al = __expf(mold - mnew);
      float p0 = __expf(t0 - mnew);
      float p1 = __expf(t1 - mnew);
      float rs = p0 + p1;
#pragma unroll
      for (int msk = 8; msk; msk >>= 1)
        rs += __shfl_xor(rs, msk, 32);
      if (n == 0) { Mst[w][row] = mnew; Lst[w][row] = lold * al + rs; }
      alpha[e] = al;
      Ps[w][row * 32 + n]      = (__bf16)p0;         // redistribute P to A layout
      Ps[w][row * 32 + 16 + n] = (__bf16)p1;
    }
#pragma unroll
    for (int e = 0; e < 8; ++e) {                    // rescale running output
      o0[e] *= alpha[e]; o1[e] *= alpha[e]; o2[e] *= alpha[e]; o3[e] *= alpha[e];
    }

    // O += P(16x32) @ V(32x64), four 16-wide d-tiles
    bf16x16 pa = load_a_frag(Ps[w], 0, 32, 0);
    o0 = wmma_bf16(pa, load_b_frag(Vts[p], 0, 32, 0), o0);
    o1 = wmma_bf16(pa, load_b_frag(Vts[p], 16, 32, 0), o1);
    o2 = wmma_bf16(pa, load_b_frag(Vts[p], 32, 32, 0), o2);
    o3 = wmma_bf16(pa, load_b_frag(Vts[p], 48, 32, 0), o3);
  }

  // finalize: divide by softmax denominator, write [B,T,C] with head offset
#pragma unroll
  for (int e = 0; e < 8; ++e) {
    int row = e + 8 * half;
    float linv = 1.0f / Lst[w][row];
    int qg = q0 + w * 16 + row;
    int ob = (b * 2048 + qg) * 1024 + h * 64 + n;
    out[ob + 0]  = o0[e] * linv;
    out[ob + 16] = o1[e] * linv;
    out[ob + 32] = o2[e] * linv;
    out[ob + 48] = o3[e] * linv;
  }
}

extern "C" void kernel_launch(void* const* d_in, const int* in_sizes, int n_in,
                              void* d_out, int out_size, void* d_ws, size_t ws_size,
                              hipStream_t stream) {
  (void)in_sizes; (void)n_in; (void)out_size; (void)ws_size;
  const float* x     = (const float*)d_in[0];   // [2,2048,1024]
  const float* w_qkv = (const float*)d_in[1];   // [1024,3072]
  const float* b_qkv = (const float*)d_in[2];   // [3072]
  const float* w_out = (const float*)d_in[3];   // [1024,1024]
  const float* b_out = (const float*)d_in[4];   // [1024]
  float* out = (float*)d_out;                   // [2,2048,1024]

  float* qkv  = (float*)d_ws;                          // 4096*3072 f32
  float* attn = qkv + (size_t)4096 * 3072;             // 4096*1024 f32

  // 1) qkv = x @ w_qkv + b_qkv          (M=4096, N=3072, K=1024)
  dim3 g1(3072 / 128, 4096 / 128);
  gemm_bias_kernel<<<g1, 256, 0, stream>>>(x, w_qkv, b_qkv, qkv, 4096, 3072, 1024);

  // 2) causal flash attention over qkv  -> attn [4096,1024]
  dim3 g2(2048 / 64, 16, 2);
  attn_kernel<<<g2, 128, 0, stream>>>(qkv, attn);

  // 3) out = attn @ w_out + b_out       (M=4096, N=1024, K=1024)
  dim3 g3(1024 / 128, 4096 / 128);
  gemm_bias_kernel<<<g3, 256, 0, stream>>>(attn, w_out, b_out, out, 4096, 1024, 1024);
}
